// IncrementalVGG16_38001870635790
// MI455X (gfx1250) — matrix operand order repaired and checked by reading, not compiled, not run on hardware
//
#include <hip/hip_runtime.h>

typedef __bf16 bf16;
typedef __attribute__((ext_vector_type(16))) __bf16 v16bf;
typedef __attribute__((ext_vector_type(8)))  __bf16 v8bf;
typedef __attribute__((ext_vector_type(8)))  float  v8f;

union V16 { v16bf v; v8bf h[2]; };

// ---------------------------------------------------------------------------
// Static per-layer incremental-patch geometry (patch 32x32, BETA=1.0, S=224).
// ---------------------------------------------------------------------------
#define NLAYERS 18
__device__ __constant__ int d_isconv[NLAYERS] = {1,1,0,1,1,0,1,1,1,0,1,1,1,0,1,1,1,0};
__device__ __constant__ int d_sizes [NLAYERS] = {224,224,112,112,112,56,56,56,56,28,28,28,28,14,14,14,14,7};
__device__ __constant__ int d_nphs  [NLAYERS] = {34,36,19,21,23,12,14,16,18,10,12,14,16,9,11,13,14,7};
__device__ __constant__ int d_fulls [NLAYERS] = {0,0,0,0,0,0,0,0,0,0,0,0,0,0,0,0,1,1};

// ---------------------------------------------------------------------------
// Per-batch patch location chain: ly/lx for all 18 layers (B=8 threads).
// ---------------------------------------------------------------------------
__global__ void compute_locs(const int* __restrict__ patch_loc,
                             int* __restrict__ lyArr, int* __restrict__ lxArr, int B_) {
  int b = threadIdx.x;
  if (b >= B_) return;
  int ly = patch_loc[2 * b + 0];
  int lx = patch_loc[2 * b + 1];
  for (int i = 0; i < NLAYERS; ++i) {
    int S = d_sizes[i], nph = d_nphs[i];
    if (d_fulls[i]) { ly = 0; lx = 0; }
    else if (d_isconv[i]) {
      int t = ly - 1; t = t < 0 ? 0 : t; t = t > S - nph ? S - nph : t; ly = t;
      t = lx - 1; t = t < 0 ? 0 : t; t = t > S - nph ? S - nph : t; lx = t;
    } else {
      int t = ly >> 1; t = t > S - nph ? S - nph : t; ly = t;
      t = lx >> 1; t = t > S - nph ? S - nph : t; lx = t;
    }
    lyArr[i * 8 + b] = ly;
    lxArr[i * 8 + b] = lx;
  }
}

// ---------------------------------------------------------------------------
// Layer-0 im2col: fp32 NCHW [8,3,H,W] -> bf16 col[pix*32 + k],
// k = (kh*3+kw)*3 + cin for k<27 (SAME padding -> zeros), k=27..31 zero.
// ---------------------------------------------------------------------------
__global__ void im2col_l0(const float* __restrict__ xNCHW, bf16* __restrict__ col,
                          int B_, int H, int W) {
  unsigned total = (unsigned)B_ * H * W * 32;
  unsigned i = blockIdx.x * blockDim.x + threadIdx.x;
  if (i >= total) return;
  int k = (int)(i & 31);
  unsigned pix = i >> 5;
  int w = (int)(pix % W); unsigned t = pix / W;
  int h = (int)(t % H); int b = (int)(t / H);
  bf16 v = (bf16)0.f;
  if (k < 27) {
    int p = k / 3, cin = k - p * 3;
    int kh = p / 3, kw = p - kh * 3;
    int ih = h + kh - 1, iw = w + kw - 1;
    if ((unsigned)ih < (unsigned)H && (unsigned)iw < (unsigned)W)
      v = (bf16)xNCHW[(((unsigned)b * 3 + cin) * H + ih) * W + iw];
  }
  col[i] = v;
}

// ---------------------------------------------------------------------------
// Conv weights [Cout][Cin][3][3] fp32 -> WMMA **A-fragment** swizzled bf16:
//   wA[((ntile*NCH + chunk)*32 + lane)*16 + e]
// A element (lane,e): M(row)=cout = ntile*16 + (lane&15),
//   K = chunk*32 + 16*grp + e, grp=lane>>4,
//   K indexes (kh,kw,cin) as k=(kh*3+kw)*Cin+cin. Out-of-range k zero-filled.
// ---------------------------------------------------------------------------
__global__ void convw_swizzleA_bf16(const float* __restrict__ w, bf16* __restrict__ wA,
                                    int Cout, int Cin, int nChunks) {
  int K = 9 * Cin;
  unsigned total = (unsigned)(Cout / 16) * nChunks * 32 * 16;
  unsigned i = blockIdx.x * blockDim.x + threadIdx.x;
  if (i >= total) return;
  int e    = (int)(i & 15);
  int lane = (int)((i >> 4) & 31);
  unsigned t = i >> 9;
  int ch = (int)(t % nChunks);
  int tn = (int)(t / nChunks);
  int grp = lane >> 4, mrow = lane & 15;
  int k = ch * 32 + 16 * grp + e;
  int n = tn * 16 + mrow;
  bf16 v = (bf16)0.f;
  if (k < K) {
    int p = k / Cin, cin = k - p * Cin;
    int kh = p / 3, kw = p - kh * 3;
    v = (bf16)w[(((unsigned)n * Cin + cin) * 3 + kh) * 3 + kw];
  }
  wA[i] = v;
}

// ---------------------------------------------------------------------------
// Shared epilogue: bias + ReLU + one b128 store per pixel tile per lane.
// ---------------------------------------------------------------------------
__device__ __forceinline__
void conv_epilogue(const v8f* accs, const float* bias, bf16* out,
                   int cout0, int grp, int mrow, unsigned pixRow,
                   int w0, int W, int Cout) {
  const float4* bp4 = (const float4*)(bias + cout0 + 8 * grp);
  const float4 bv0 = bp4[0], bv1 = bp4[1];
  const float bl[8] = {bv0.x, bv0.y, bv0.z, bv0.w, bv1.x, bv1.y, bv1.z, bv1.w};
#pragma unroll
  for (int j = 0; j < 4; ++j) {
    const int w = w0 + j * 16 + mrow;
    if (w < W) {
      v8bf o;
#pragma unroll
      for (int r = 0; r < 8; ++r) {
        float v = accs[j][r] + bl[r];
        v = v > 0.f ? v : 0.f;
        o[r] = (bf16)v;
      }
      *(v8bf*)(out + (pixRow + (unsigned)w) * (unsigned)Cout + cout0 + 8 * grp) = o;
    }
  }
}

// ---------------------------------------------------------------------------
// Layer-0 conv from im2col: single K=32 chunk, pure b128 + wmma path.
// ---------------------------------------------------------------------------
__global__ __launch_bounds__(32)
void conv_l0_wmma(const bf16* __restrict__ col, const bf16* __restrict__ wA,
                  const float* __restrict__ bias, bf16* __restrict__ out,
                  int H, int W, int Cout) {
  const int lane = threadIdx.x;
  const int grp  = lane >> 4;
  const int mrow = lane & 15;
  const int PT   = (W + 63) >> 6;
  int bx = blockIdx.x;
  const int pt = bx % PT; bx /= PT;
  const int h  = bx % H;
  const int b  = bx / H;
  const int w0 = pt * 64;
  const int tn = blockIdx.y;
  const int cout0 = tn * 16;

  const bf16* ap = wA + ((unsigned)tn * 32 + lane) * 16;
  V16 a;
  a.h[0] = *(const v8bf*)(ap);
  a.h[1] = *(const v8bf*)(ap + 8);

  const unsigned pixRow = ((unsigned)b * H + (unsigned)h) * W;
  v8f acc0 = {}, acc1 = {}, acc2 = {}, acc3 = {};
#pragma unroll
  for (int j = 0; j < 4; ++j) {
    const int w = w0 + j * 16 + mrow;
    V16 bm;
    if (w < W) {
      const bf16* pp = col + (pixRow + (unsigned)w) * 32u + 16 * grp;
      bm.h[0] = *(const v8bf*)(pp);
      bm.h[1] = *(const v8bf*)(pp + 8);
    } else {
      v8bf z = {}; bm.h[0] = z; bm.h[1] = z;
    }
    if (j == 0) acc0 = __builtin_amdgcn_wmma_f32_16x16x32_bf16(false, a.v, false, bm.v, (short)0, acc0, false, false);
    if (j == 1) acc1 = __builtin_amdgcn_wmma_f32_16x16x32_bf16(false, a.v, false, bm.v, (short)0, acc1, false, false);
    if (j == 2) acc2 = __builtin_amdgcn_wmma_f32_16x16x32_bf16(false, a.v, false, bm.v, (short)0, acc2, false, false);
    if (j == 3) acc3 = __builtin_amdgcn_wmma_f32_16x16x32_bf16(false, a.v, false, bm.v, (short)0, acc3, false, false);
  }
  v8f accs[4] = {acc0, acc1, acc2, acc3};
  conv_epilogue(accs, bias, out, cout0, grp, mrow, pixRow, w0, W, Cout);
}

// ---------------------------------------------------------------------------
// 3x3 SAME conv + bias + ReLU, implicit GEMM, M=cout N=pixels orientation.
// CIN % 32 == 0 (layers 2..13). One wave32 per 16(cout) x 64(pixels).
// ---------------------------------------------------------------------------
template<int CIN>
__global__ __launch_bounds__(32)
void conv3x3_wmma(const bf16* __restrict__ in, const bf16* __restrict__ wA,
                  const float* __restrict__ bias, bf16* __restrict__ out,
                  int H, int W, int Cout) {
  static_assert(CIN % 32 == 0, "fast path requires CIN % 32 == 0");
  constexpr int K   = 9 * CIN;
  constexpr int NCH = K / 32;
  const int lane = threadIdx.x;
  const int grp  = lane >> 4;
  const int mrow = lane & 15;
  const int PT   = (W + 63) >> 6;
  int bx = blockIdx.x;
  const int pt = bx % PT; bx /= PT;
  const int h  = bx % H;
  const int b  = bx / H;
  const int w0 = pt * 64;
  const int tn = blockIdx.y;
  const int cout0 = tn * 16;

  v8f acc0 = {}, acc1 = {}, acc2 = {}, acc3 = {};

  for (int ch = 0; ch < NCH; ++ch) {
    // ---- A operand: swizzled weights, 2 coalesced b128 loads ----
    const bf16* ap = wA + (((unsigned)tn * NCH + ch) * 32 + lane) * 16;
    if (ch + 1 < NCH) __builtin_prefetch(ap + 32 * 16, 0, 0);
    V16 a;
    a.h[0] = *(const v8bf*)(ap);
    a.h[1] = *(const v8bf*)(ap + 8);

    // ---- B operand: 4 pixel tiles, 2 contiguous b128 loads each ----
    const int k0 = ch * 32;
    const int p = k0 / CIN, cin0 = k0 - p * CIN;
    const int kh = p / 3, kw = p - kh * 3;
    const int ih = h + kh - 1;
    const bool rowok = (unsigned)ih < (unsigned)H;
    const unsigned rowBase = ((unsigned)b * H + (unsigned)ih) * W;
    const int cofs = cin0 + 16 * grp;
#pragma unroll
    for (int j = 0; j < 4; ++j) {
      const int iw = w0 + j * 16 + mrow + kw - 1;
      V16 bm;
      if (rowok && (unsigned)iw < (unsigned)W) {
        const bf16* pp = in + (rowBase + (unsigned)iw) * (unsigned)CIN + cofs;
        bm.h[0] = *(const v8bf*)(pp);
        bm.h[1] = *(const v8bf*)(pp + 8);
      } else {
        v8bf z = {}; bm.h[0] = z; bm.h[1] = z;
      }
      if (j == 0) acc0 = __builtin_amdgcn_wmma_f32_16x16x32_bf16(false, a.v, false, bm.v, (short)0, acc0, false, false);
      if (j == 1) acc1 = __builtin_amdgcn_wmma_f32_16x16x32_bf16(false, a.v, false, bm.v, (short)0, acc1, false, false);
      if (j == 2) acc2 = __builtin_amdgcn_wmma_f32_16x16x32_bf16(false, a.v, false, bm.v, (short)0, acc2, false, false);
      if (j == 3) acc3 = __builtin_amdgcn_wmma_f32_16x16x32_bf16(false, a.v, false, bm.v, (short)0, acc3, false, false);
    }
  }

  v8f accs[4] = {acc0, acc1, acc2, acc3};
  conv_epilogue(accs, bias, out,
                cout0, grp, mrow, ((unsigned)b * H + (unsigned)h) * W, w0, W, Cout);
}

// ---------------------------------------------------------------------------
// 2x2 stride-2 maxpool, bf16 NHWC, vectorized x8 (C % 8 == 0)
// ---------------------------------------------------------------------------
__global__ void maxpool2_nhwc_v8(const bf16* __restrict__ in, bf16* __restrict__ out,
                                 int B_, int Hi, int Wi, int C) {
  int Ho = Hi >> 1, Wo = Wi >> 1;
  int G = C >> 3;
  unsigned total = (unsigned)B_ * Ho * Wo * G;
  unsigned i = blockIdx.x * blockDim.x + threadIdx.x;
  if (i >= total) return;
  int g = (int)(i % G); unsigned t = i / G;
  int w = (int)(t % Wo); t /= Wo;
  int h = (int)(t % Ho); int b = (int)(t / Ho);
  unsigned base = (((unsigned)b * Hi + 2 * h) * Wi + 2 * w) * (unsigned)C + (unsigned)g * 8;
  v8bf p0 = *(const v8bf*)(in + base);
  v8bf p1 = *(const v8bf*)(in + base + C);
  v8bf p2 = *(const v8bf*)(in + base + (unsigned)Wi * C);
  v8bf p3 = *(const v8bf*)(in + base + (unsigned)Wi * C + C);
  v8bf o;
#pragma unroll
  for (int r = 0; r < 8; ++r) {
    float a = (float)p0[r], bq = (float)p1[r], c = (float)p2[r], d = (float)p3[r];
    float m01 = a > bq ? a : bq;
    float m23 = c > d ? c : d;
    o[r] = (bf16)(m01 > m23 ? m01 : m23);
  }
  *(v8bf*)(out + (size_t)i * 8) = o;
}

// ---------------------------------------------------------------------------
// Patch merge, vectorized x8: out = in-patch ? full : cache (in-place legal)
// ---------------------------------------------------------------------------
__global__ void merge_patch_v8(const bf16* __restrict__ full, const bf16* __restrict__ cache,
                               bf16* __restrict__ out,
                               const int* __restrict__ lyA, const int* __restrict__ lxA,
                               int nph, int B_, int S, int C) {
  int G = C >> 3;
  unsigned total = (unsigned)B_ * S * S * G;
  unsigned i = blockIdx.x * blockDim.x + threadIdx.x;
  if (i >= total) return;
  unsigned t = i / G;
  int w = (int)(t % S); t /= S;
  int h = (int)(t % S); int b = (int)(t / S);
  int ly = lyA[b], lx = lxA[b];
  bool inp = (h >= ly) && (h < ly + nph) && (w >= lx) && (w < lx + nph);
  v8bf v = inp ? ((const v8bf*)full)[i] : ((const v8bf*)cache)[i];
  ((v8bf*)out)[i] = v;
}

// ---------------------------------------------------------------------------
// Flatten final NHWC [8,7,7,512] map into NCHW-flat bf16 [8, 25088]
// (flat[m][c*49 + h*7 + w] = nhwc[((m*7+h)*7+w)*512 + c]).
// ---------------------------------------------------------------------------
__global__ void flatten_nchw_bf16(const bf16* __restrict__ nhwc, bf16* __restrict__ flat) {
  unsigned total = 8u * 25088u;
  unsigned i = blockIdx.x * blockDim.x + threadIdx.x;
  if (i >= total) return;
  unsigned m = i / 25088u, r = i % 25088u;
  unsigned c = r / 49u, q = r % 49u;
  unsigned hh = q / 7u, ww = q % 7u;
  flat[i] = nhwc[(((m * 7u + hh) * 7u + ww) * 512u) + c];
}

// ---------------------------------------------------------------------------
// FC GEMM via WMMA: [M<=16, K] (bf16, row-contiguous) x [K, N] (fp32 weights,
// cast in-register; lane-coalesced column gather).
// ---------------------------------------------------------------------------
template<bool RELU, bool OUTF32>
__global__ __launch_bounds__(32)
void fc_wmma(const bf16* __restrict__ A, const float* __restrict__ Wf,
             const float* __restrict__ bias, void* __restrict__ out,
             int K, int N, int Mvalid) {
  const int lane = threadIdx.x;
  const int grp  = lane >> 4;
  const int mrow = lane & 15;
  const int n0 = blockIdx.x * 16;

  v8f acc = {};
  for (int k0 = 0; k0 < K; k0 += 32) {
    V16 a;
    if (mrow < Mvalid) {
      const bf16* pr = A + (unsigned)mrow * K + k0 + 8 * grp;
      a.h[0] = *(const v8bf*)(pr);
      a.h[1] = *(const v8bf*)(pr + 16);
    } else {
      v8bf z = {}; a.h[0] = z; a.h[1] = z;
    }
    V16 bm;
#pragma unroll
    for (int e = 0; e < 16; ++e) {
      int kb = k0 + (e < 8 ? 8 * grp + e : 8 + 8 * grp + e);
      bf16 bv = (bf16)0.f;
      if ((n0 + mrow) < N) bv = (bf16)Wf[(unsigned)kb * N + n0 + mrow];
      bm.v[e] = bv;
    }
    acc = __builtin_amdgcn_wmma_f32_16x16x32_bf16(false, a.v, false, bm.v,
                                                  (short)0, acc, false, false);
  }
#pragma unroll
  for (int r = 0; r < 8; ++r) {
    int m = r + 8 * grp;
    int n = n0 + mrow;
    if (m < Mvalid && n < N) {
      float v = acc[r] + bias[n];
      if (RELU) v = v > 0.f ? v : 0.f;
      if (OUTF32) ((float*)out)[(unsigned)m * N + n] = v;
      else        ((bf16*)out)[(unsigned)m * N + n] = (bf16)v;
    }
  }
}

// ---------------------------------------------------------------------------
// Host orchestration
// ---------------------------------------------------------------------------
static inline void launch_conv(int Cin, dim3 g, dim3 blk, hipStream_t s,
                               const bf16* in, const bf16* wA, const float* bias,
                               bf16* out, int H, int W, int Cout) {
  switch (Cin) {
    case 64:  hipLaunchKernelGGL((conv3x3_wmma<64>),  g, blk, 0, s, in, wA, bias, out, H, W, Cout); break;
    case 128: hipLaunchKernelGGL((conv3x3_wmma<128>), g, blk, 0, s, in, wA, bias, out, H, W, Cout); break;
    case 256: hipLaunchKernelGGL((conv3x3_wmma<256>), g, blk, 0, s, in, wA, bias, out, H, W, Cout); break;
    default:  hipLaunchKernelGGL((conv3x3_wmma<512>), g, blk, 0, s, in, wA, bias, out, H, W, Cout); break;
  }
}

extern "C" void kernel_launch(void* const* d_in, const int* in_sizes, int n_in,
                              void* d_out, int out_size, void* d_ws, size_t ws_size,
                              hipStream_t stream) {
  (void)in_sizes; (void)n_in; (void)out_size; (void)ws_size;
  const int B_ = 8;

  const float* x    = (const float*)d_in[0];
  const float* x0   = (const float*)d_in[1];
  const int*   ploc = (const int*)d_in[2];
  // d_in[3], d_in[4] = p_height/p_width scalars (fixed 32, folded into tables)

  const float* cw[13]; const float* cb[13];
  for (int i = 0; i < 13; ++i) { cw[i] = (const float*)d_in[5 + 2 * i]; cb[i] = (const float*)d_in[6 + 2 * i]; }
  const float* fw[3]; const float* fb[3];
  for (int i = 0; i < 3; ++i) { fw[i] = (const float*)d_in[31 + 2 * i]; fb[i] = (const float*)d_in[32 + 2 * i]; }

  // ---- workspace carve (bump allocator) ----
  char* ws = (char*)d_ws; size_t off = 0;
  auto alloc = [&](size_t bytes) -> void* {
    void* p = ws + off;
    off = (off + bytes + 255) & ~(size_t)255;
    return p;
  };
  const size_t MAXACT = (size_t)B_ * 224 * 224 * 64;  // biggest bf16 activation map
  bf16* bufA = (bf16*)alloc(MAXACT * sizeof(bf16));
  bf16* bufB = (bf16*)alloc(MAXACT * sizeof(bf16));
  bf16* cacA = (bf16*)alloc(MAXACT * sizeof(bf16));
  bf16* cacB = (bf16*)alloc(MAXACT * sizeof(bf16));
  bf16* colX  = (bf16*)alloc((size_t)B_ * 224 * 224 * 32 * sizeof(bf16));
  bf16* colX0 = (bf16*)alloc((size_t)B_ * 224 * 224 * 32 * sizeof(bf16));
  bf16* flatV = (bf16*)alloc((size_t)B_ * 25088 * sizeof(bf16));
  bf16* fcA  = (bf16*)alloc((size_t)B_ * 4096 * sizeof(bf16));
  bf16* fcB  = (bf16*)alloc((size_t)B_ * 4096 * sizeof(bf16));
  int*  lyArr = (int*)alloc(NLAYERS * 8 * sizeof(int));
  int*  lxArr = (int*)alloc(NLAYERS * 8 * sizeof(int));

  static const int convCin [13] = {3,64,64,128,128,256,256,256,512,512,512,512,512};
  static const int convCout[13] = {64,64,128,128,256,256,256,512,512,512,512,512,512};
  bf16* wA[13]; int nCh[13];
  for (int i = 0; i < 13; ++i) {
    nCh[i] = (9 * convCin[i] + 31) / 32;
    wA[i] = (bf16*)alloc((size_t)(convCout[i] / 16) * nCh[i] * 32 * 16 * sizeof(bf16));
  }

  // ---- per-batch patch location chains ----
  hipLaunchKernelGGL(compute_locs, dim3(1), dim3(8), 0, stream, ploc, lyArr, lxArr, B_);

  // ---- weight conversion into WMMA-A-swizzled bf16 ----
  for (int i = 0; i < 13; ++i) {
    size_t tot = (size_t)(convCout[i] / 16) * nCh[i] * 32 * 16;
    hipLaunchKernelGGL(convw_swizzleA_bf16, dim3((unsigned)((tot + 255) / 256)), dim3(256), 0, stream,
                       cw[i], wA[i], convCout[i], convCin[i], nCh[i]);
  }

  // ---- layer-0 im2col (fp32 NCHW -> padded K=32 bf16 col) ----
  {
    size_t tot = (size_t)B_ * 224 * 224 * 32;
    unsigned g = (unsigned)((tot + 255) / 256);
    hipLaunchKernelGGL(im2col_l0, dim3(g), dim3(256), 0, stream, x,  colX,  B_, 224, 224);
    hipLaunchKernelGGL(im2col_l0, dim3(g), dim3(256), 0, stream, x0, colX0, B_, 224, 224);
  }

  // ---- 18-layer incremental + cache chains in lockstep ----
  static const int isconv[NLAYERS] = {1,1,0,1,1,0,1,1,1,0,1,1,1,0,1,1,1,0};
  static const int nphs  [NLAYERS] = {34,36,19,21,23,12,14,16,18,10,12,14,16,9,11,13,14,7};
  static const int fulls [NLAYERS] = {0,0,0,0,0,0,0,0,0,0,0,0,0,0,0,0,1,1};

  bf16 *bufIn = bufA, *bufOut = bufB, *cacIn = cacA, *cacOut = cacB;
  int curS = 224, curC = 3, ci = 0;

  for (int li = 0; li < NLAYERS; ++li) {
    if (isconv[li]) {
      int Cin = convCin[ci], Cout = convCout[ci];
      int H = curS, W = curS;
      int PT = (W + 63) >> 6;
      dim3 g((unsigned)(B_ * H * PT), (unsigned)(Cout / 16));
      dim3 blk(32);
      if (li == 0) {
        hipLaunchKernelGGL(conv_l0_wmma, g, blk, 0, stream, colX0, wA[0], cb[0], cacOut, H, W, Cout);
        hipLaunchKernelGGL(conv_l0_wmma, g, blk, 0, stream, colX,  wA[0], cb[0], bufOut, H, W, Cout);
      } else {
        if (!fulls[li])  // cache chain not consumed after full-recompute layers
          launch_conv(Cin, g, blk, stream, cacIn, wA[ci], cb[ci], cacOut, H, W, Cout);
        launch_conv(Cin, g, blk, stream, bufIn, wA[ci], cb[ci], bufOut, H, W, Cout);
      }
      curC = Cout; ci++;
    } else {
      int Hi = curS, Wi = curS;
      curS >>= 1;
      size_t tot = (size_t)B_ * curS * curS * (curC >> 3);
      unsigned gp = (unsigned)((tot + 255) / 256);
      if (!fulls[li])
        hipLaunchKernelGGL(maxpool2_nhwc_v8, dim3(gp), dim3(256), 0, stream, cacIn, cacOut, B_, Hi, Wi, curC);
      hipLaunchKernelGGL(maxpool2_nhwc_v8, dim3(gp), dim3(256), 0, stream, bufIn, bufOut, B_, Hi, Wi, curC);
    }
    if (!fulls[li]) {
      size_t tot = (size_t)B_ * curS * curS * (curC >> 3);
      unsigned gm = (unsigned)((tot + 255) / 256);
      hipLaunchKernelGGL(merge_patch_v8, dim3(gm), dim3(256), 0, stream,
                         bufOut, cacOut, bufOut, lyArr + li * 8, lxArr + li * 8,
                         nphs[li], B_, curS, curC);
      bf16* t = cacIn; cacIn = cacOut; cacOut = t;
    }
    bf16* t = bufIn; bufIn = bufOut; bufOut = t;
  }
  // bufIn now holds final feature map: bf16 NHWC [8,7,7,512]

  // ---- FC head ----
  hipLaunchKernelGGL(flatten_nchw_bf16, dim3((8 * 25088 + 255) / 256), dim3(256), 0, stream,
                     bufIn, flatV);
  hipLaunchKernelGGL((fc_wmma<true,  false>), dim3(4096 / 16), dim3(32), 0, stream,
                     flatV, fw[0], fb[0], fcA, 512 * 7 * 7, 4096, B_);
  hipLaunchKernelGGL((fc_wmma<true,  false>), dim3(4096 / 16), dim3(32), 0, stream,
                     fcA, fw[1], fb[1], fcB, 4096, 4096, B_);
  hipLaunchKernelGGL((fc_wmma<false, true>),  dim3((1000 + 15) / 16), dim3(32), 0, stream,
                     fcB, fw[2], fb[2], d_out, 4096, 1000, B_);
}
